// Mask2FormerDecoderLayer_71305047048435
// MI455X (gfx1250) — matrix operand order repaired and checked by reading, not compiled, not run on hardware
//
#include <hip/hip_runtime.h>

// ---------------------------------------------------------------------------
// Mask2Former decoder layer forward, bf16 WMMA pipeline for gfx1250 (MI455X)
// ---------------------------------------------------------------------------

typedef __bf16 bf16_t;
typedef __attribute__((ext_vector_type(16))) __bf16 v16bf;
typedef __attribute__((ext_vector_type(8)))  float  v8f;

#define Bsz    16
#define NQ     300
#define Cdim   256
#define Hh     8
#define DFF    2048
#define DH     32
#define LEN_IN 21504
#define NQP    320              // NQ padded to multiple of 32 for WMMA K/M/N
#define Mrows  (Bsz * NQ)       // 4800
#define BH     (Bsz * Hh)       // 128

#define GF_A_F32    1
#define GF_RELU     2
#define GF_OUT_BF16 4
#define GF_BIAS     8
#define GF_MASK     16

// ---------------------------------------------------------------------------
// Generic WMMA GEMM:  C[z] = A[z] (MxK) * B[z]^T (B stored [N,K]) + bias
// Wave computes a 16x(16*NT) tile; 8 waves per block stacked in M.
// B rows for the full 16*NT tile must be loadable (pad weights if needed);
// the N bound is enforced only at the store epilogue.
// A operand: lane m=lane%16, chunks K+koff..+7 and K+koff+16..+23 (koff=8*(lane>=16))
// B operand: lane n=lane%16, chunks K+koff..+7 and K+koff+16..+23 (koff=8*(lane>=16))
// D: vgpr r -> m = r + 8*(lane>=16), n = lane%16
// ---------------------------------------------------------------------------
template <int NT, int FLAGS>
__global__ __launch_bounds__(256)
void wmma_gemm_kernel(const void* __restrict__ Av, int lda, long long sA,
                      const bf16_t* __restrict__ Bw, int ldb, long long sB,
                      const float* __restrict__ bias,
                      void* __restrict__ Cv, int ldc, long long sC,
                      int M, int N, int K,
                      const unsigned char* __restrict__ mask)
{
    const int lane = threadIdx.x & 31;
    const int wave = threadIdx.x >> 5;
    const int m0   = (blockIdx.y * 8 + wave) * 16;
    if (m0 + 16 > M) return;
    const int    n0 = blockIdx.x * (16 * NT);
    const size_t z  = blockIdx.z;

    const int mrow   = m0 + (lane & 15);
    const int a_koff = (lane >> 4) * 8;
    const int b_koff = (lane >> 4) * 8;
    const int ncol   = lane & 15;

    const bf16_t* Ab = (const bf16_t*)Av + z * (size_t)sA + (size_t)mrow * lda + a_koff;
    const float*  Af = (const float* )Av + z * (size_t)sA + (size_t)mrow * lda + a_koff;

    const bf16_t* Bz = Bw + z * (size_t)sB;
    const bf16_t* Brow[NT];
#pragma unroll
    for (int j = 0; j < NT; ++j)
        Brow[j] = Bz + (size_t)(n0 + j * 16 + ncol) * ldb + b_koff;

    v8f acc[NT] = {};

#pragma unroll 2
    for (int k = 0; k < K; k += 32) {
        v16bf a;
        if constexpr (FLAGS & GF_A_F32) {
            const float* p = Af + k;
            float4 f0 = *(const float4*)(p);
            float4 f1 = *(const float4*)(p + 4);
            float4 f2 = *(const float4*)(p + 16);
            float4 f3 = *(const float4*)(p + 20);
            a[0]=(__bf16)f0.x;  a[1]=(__bf16)f0.y;  a[2]=(__bf16)f0.z;  a[3]=(__bf16)f0.w;
            a[4]=(__bf16)f1.x;  a[5]=(__bf16)f1.y;  a[6]=(__bf16)f1.z;  a[7]=(__bf16)f1.w;
            a[8]=(__bf16)f2.x;  a[9]=(__bf16)f2.y;  a[10]=(__bf16)f2.z; a[11]=(__bf16)f2.w;
            a[12]=(__bf16)f3.x; a[13]=(__bf16)f3.y; a[14]=(__bf16)f3.z; a[15]=(__bf16)f3.w;
        } else {
            const bf16_t* p = Ab + k;
            union { uint4 u[2]; v16bf v; } t;
            t.u[0] = *(const uint4*)(p);
            t.u[1] = *(const uint4*)(p + 16);
            a = t.v;
        }
        v16bf b[NT];
#pragma unroll
        for (int j = 0; j < NT; ++j) {
            union { uint4 u[2]; v16bf v; } tb;
            const bf16_t* bp = Brow[j] + k;
            tb.u[0] = *(const uint4*)(bp);
            tb.u[1] = *(const uint4*)(bp + 16);
            b[j] = tb.v;
        }
#pragma unroll
        for (int j = 0; j < NT; ++j)
            acc[j] = __builtin_amdgcn_wmma_f32_16x16x32_bf16(
                false, a, false, b[j], (short)0, acc[j], false, false);
    }

    const int mbase = m0 + ((lane >> 4) << 3);
#pragma unroll
    for (int j = 0; j < NT; ++j) {
        const int n = n0 + j * 16 + ncol;
        if (n0 + j * 16 + 16 > N) continue;   // epilogue-only bound (B rows padded)
        float bv = 0.f;
        if constexpr (FLAGS & GF_BIAS) bv = bias[n];
#pragma unroll
        for (int r = 0; r < 8; ++r) {
            const int m = mbase + r;
            float v = acc[j][r] + bv;
            if constexpr (FLAGS & GF_RELU) v = fmaxf(v, 0.f);
            if constexpr (FLAGS & GF_MASK) { if (mask[m]) v = 0.f; }
            const size_t o = z * (size_t)sC + (size_t)m * ldc + n;
            if constexpr (FLAGS & GF_OUT_BF16) ((bf16_t*)Cv)[o] = (bf16_t)v;
            else                               ((float* )Cv)[o] = v;
        }
    }
}

// LayerNorm over C=256; out1 = bf16(LN(x)); out2 = bf16(LN(x)+pos) (optional)
__global__ __launch_bounds__(256)
void ln_kernel(const float* __restrict__ x, const float* __restrict__ g,
               const float* __restrict__ bta, const float* __restrict__ pos,
               bf16_t* __restrict__ out1, bf16_t* __restrict__ out2)
{
    const int row = blockIdx.x;
    const int t   = threadIdx.x;
    __shared__ float red[256];
    const size_t base = (size_t)row * Cdim;
    const float  v = x[base + t];
    red[t] = v; __syncthreads();
    for (int s = 128; s > 0; s >>= 1) { if (t < s) red[t] += red[t + s]; __syncthreads(); }
    const float mu = red[0] * (1.f / 256.f);
    __syncthreads();
    const float d = v - mu;
    red[t] = d * d; __syncthreads();
    for (int s = 128; s > 0; s >>= 1) { if (t < s) red[t] += red[t + s]; __syncthreads(); }
    const float inv = rsqrtf(red[0] * (1.f / 256.f) + 1e-5f);
    const float y   = d * inv * g[t] + bta[t];
    out1[base + t] = (bf16_t)y;
    if (out2) out2[base + t] = (bf16_t)(y + pos[base + t]);
}

__global__ void cvt_f32_bf16_kernel(const float* __restrict__ s, bf16_t* __restrict__ d, int n)
{
    int i = blockIdx.x * blockDim.x + threadIdx.x;
    if (i < n) d[i] = (bf16_t)s[i];
}

__global__ void zero_bf16_kernel(bf16_t* __restrict__ p, int n)
{
    int i = blockIdx.x * blockDim.x + threadIdx.x;
    if (i < n) p[i] = (bf16_t)0.f;
}

__global__ void add_f32_kernel(const float* __restrict__ a, const float* __restrict__ b,
                               float* __restrict__ o, int n)
{
    int i = blockIdx.x * blockDim.x + threadIdx.x;
    if (i < n) o[i] = a[i] + b[i];
}

// qkv f32 [4800,768] -> Q,K bf16 [B,H,NQP,DH] (Q pre-scaled by 1/sqrt(DH)),
// Vt bf16 [B,H,DH,NQP]; pads were pre-zeroed.
__global__ void split_qkv_kernel(const float* __restrict__ qkv,
                                 bf16_t* __restrict__ Q, bf16_t* __restrict__ Kb,
                                 bf16_t* __restrict__ Vt)
{
    int gid = blockIdx.x * blockDim.x + threadIdx.x;
    if (gid >= Mrows * Cdim) return;
    int mrow = gid >> 8;
    int cc   = gid & 255;
    int h = cc >> 5, d = cc & 31;
    int b = mrow / NQ, q = mrow % NQ;
    const float* r = qkv + (size_t)mrow * 768;
    size_t bh = (size_t)(b * Hh + h);
    Q [(bh * NQP + q) * DH + d]  = (bf16_t)(r[cc] * 0.17677669529663687f);
    Kb[(bh * NQP + q) * DH + d]  = (bf16_t)r[256 + cc];
    Vt[(bh * DH  + d) * NQP + q] = (bf16_t)r[512 + cc];
}

// softmax over k (valid 0..299) for rows [BH*NQP]; one wave per row (wave32)
__global__ __launch_bounds__(256)
void attn_softmax_kernel(const float* __restrict__ sc, bf16_t* __restrict__ at)
{
    const int w    = blockIdx.x * 8 + (threadIdx.x >> 5);
    const int lane = threadIdx.x & 31;
    const int q    = w % NQP;
    const float* row  = sc + (size_t)w * NQP;
    bf16_t*      orow = at + (size_t)w * NQP;
    if (q >= NQ) {
        for (int k = lane; k < NQP; k += 32) orow[k] = (bf16_t)0.f;
        return;
    }
    float e[10];
    float m = -1e30f;
#pragma unroll
    for (int i = 0; i < 10; ++i) {
        int k = lane + i * 32;
        float s = (k < NQ) ? row[k] : -1e30f;
        e[i] = s;
        m = fmaxf(m, s);
    }
    for (int o = 16; o > 0; o >>= 1) m = fmaxf(m, __shfl_xor(m, o, 32));
    float sum = 0.f;
#pragma unroll
    for (int i = 0; i < 10; ++i) {
        int k = lane + i * 32;
        float ee = (k < NQ) ? __expf(e[i] - m) : 0.f;
        e[i] = ee;
        sum += ee;
    }
    for (int o = 16; o > 0; o >>= 1) sum += __shfl_xor(sum, o, 32);
    const float inv = 1.f / sum;
#pragma unroll
    for (int i = 0; i < 10; ++i) {
        int k = lane + i * 32;
        orow[k] = (bf16_t)(e[i] * inv);
    }
}

// sa f32 [B,H,NQP,DH] -> bf16 [Mrows, C]
__global__ void merge_sa_kernel(const float* __restrict__ sa, bf16_t* __restrict__ out)
{
    int gid = blockIdx.x * blockDim.x + threadIdx.x;
    if (gid >= Mrows * Cdim) return;
    int mrow = gid >> 8;
    int cc   = gid & 255;
    int h = cc >> 5, d = cc & 31;
    int b = mrow / NQ, q = mrow % NQ;
    out[gid] = (bf16_t)sa[(((size_t)(b * Hh + h)) * NQP + q) * DH + d];
}

// softmax over L*P = 12 sampling weights per (row, head)
__global__ void aw_softmax_kernel(const float* __restrict__ aw, float* __restrict__ out)
{
    int i = blockIdx.x * blockDim.x + threadIdx.x;
    if (i >= Mrows * Hh) return;
    const float* r = aw  + (size_t)(i / Hh) * 96 + (i % Hh) * 12;
    float*       o = out + (size_t)(i / Hh) * 96 + (i % Hh) * 12;
    float mx = -1e30f;
#pragma unroll
    for (int j = 0; j < 12; ++j) mx = fmaxf(mx, r[j]);
    float e[12], s = 0.f;
#pragma unroll
    for (int j = 0; j < 12; ++j) { e[j] = __expf(r[j] - mx); s += e[j]; }
    const float inv = 1.f / s;
#pragma unroll
    for (int j = 0; j < 12; ++j) o[j] = e[j] * inv;
}

// Multi-scale deformable bilinear sampling. Wave per (query-row, head), lane = channel d.
__global__ __launch_bounds__(256)
void deform_sample_kernel(const float* __restrict__ off, const float* __restrict__ awsm,
                          const float* __restrict__ qrp, const bf16_t* __restrict__ vbf,
                          bf16_t* __restrict__ out)
{
    const int m = blockIdx.x;
    const int h = threadIdx.x >> 5;
    const int d = threadIdx.x & 31;
    const int b = m / NQ;
    const int HW[3] = {128, 64, 32};
    const int ST[3] = {0, 16384, 20480};
    float acc = 0.f;
#pragma unroll
    for (int l = 0; l < 3; ++l) {
        const int   wl = HW[l], hl = HW[l], st = ST[l];
        const float Wf = (float)wl, Hf = (float)hl;
        const float rx = qrp[(m * 3 + l) * 2 + 0];
        const float ry = qrp[(m * 3 + l) * 2 + 1];
#pragma unroll
        for (int p = 0; p < 4; ++p) {
            const int ob = m * 192 + ((h * 3 + l) * 4 + p) * 2;
            const float x = (rx + off[ob + 0] / Wf) * Wf - 0.5f;
            const float y = (ry + off[ob + 1] / Hf) * Hf - 0.5f;
            const float x0f = floorf(x), y0f = floorf(y);
            const float fx = x - x0f, fy = y - y0f;
            const int ix = (int)x0f, iy = (int)y0f;
            const float ap = awsm[m * 96 + (h * 3 + l) * 4 + p];
#pragma unroll
            for (int dy = 0; dy < 2; ++dy) {
#pragma unroll
                for (int dx = 0; dx < 2; ++dx) {
                    const int xi = ix + dx, yi = iy + dy;
                    const float w = (dx ? fx : 1.f - fx) * (dy ? fy : 1.f - fy);
                    const bool valid = (xi >= 0) && (xi < wl) && (yi >= 0) && (yi < hl);
                    const int cx = min(max(xi, 0), wl - 1);
                    const int cy = min(max(yi, 0), hl - 1);
                    const int idx = st + cy * wl + cx;
                    const float val = (float)vbf[(((size_t)b * LEN_IN + idx) * Hh + h) * DH + d];
                    if (valid) acc += ap * w * val;
                }
            }
        }
    }
    out[(size_t)m * Cdim + h * DH + d] = (bf16_t)acc;
}

// ---------------------------------------------------------------------------
template <int NT, int FLAGS>
static inline void launch_gemm(hipStream_t s, const void* A, int lda, long long sA,
                               const bf16_t* Bw, int ldb, long long sB,
                               const float* bias, void* C, int ldc, long long sC,
                               int M, int N, int K,
                               const unsigned char* mask, int batch)
{
    dim3 g((N + 16 * NT - 1) / (16 * NT), (M / 16 + 7) / 8, batch);
    wmma_gemm_kernel<NT, FLAGS><<<g, 256, 0, s>>>(A, lda, sA, Bw, ldb, sB, bias,
                                                  C, ldc, sC, M, N, K, mask);
}

extern "C" void kernel_launch(void* const* d_in, const int* in_sizes, int n_in,
                              void* d_out, int out_size, void* d_ws, size_t ws_size,
                              hipStream_t stream)
{
    (void)in_sizes; (void)n_in; (void)out_size; (void)ws_size;

    const float* tgt  = (const float*)d_in[0];
    const float* mem  = (const float*)d_in[1];
    /* d_in[2] value: unused by reference (v recomputed from memory) */
    const float* qpos = (const float*)d_in[3];
    /* d_in[4] key_pos unused, d_in[6] key_ref_points unused, d_in[8] spatial hardcoded */
    const float* qrp  = (const float*)d_in[5];
    const unsigned char* kmask = (const unsigned char*)d_in[7];
    const float* ln1g = (const float*)d_in[9],  *ln1b = (const float*)d_in[10];
    const float* ln2g = (const float*)d_in[11], *ln2b = (const float*)d_in[12];
    const float* ln3g = (const float*)d_in[13], *ln3b = (const float*)d_in[14];
    const float* Wqkv = (const float*)d_in[15], *bqkv = (const float*)d_in[16];
    const float* Wout = (const float*)d_in[17], *bout = (const float*)d_in[18];
    const float* Woff = (const float*)d_in[19], *boff = (const float*)d_in[20];
    const float* Waw  = (const float*)d_in[21], *baw  = (const float*)d_in[22];
    const float* Wv   = (const float*)d_in[23], *bv   = (const float*)d_in[24];
    const float* Wo   = (const float*)d_in[25], *bo   = (const float*)d_in[26];
    const float* W1   = (const float*)d_in[27], *b1   = (const float*)d_in[28];
    const float* W2   = (const float*)d_in[29], *b2   = (const float*)d_in[30];

    char*  ws  = (char*)d_ws;
    size_t cur = 0;
    auto alloc = [&](size_t bytes) -> void* {
        void* p = ws + cur;
        cur += (bytes + 255) & ~(size_t)255;
        return p;
    };

    // bf16 weights (aw padded 96 -> 128 rows so B-loads are branch-free)
    bf16_t* wqkv_bf = (bf16_t*)alloc(768 * 256 * 2);
    bf16_t* wout_bf = (bf16_t*)alloc(256 * 256 * 2);
    bf16_t* wv_bf   = (bf16_t*)alloc(256 * 256 * 2);
    bf16_t* woff_bf = (bf16_t*)alloc(192 * 256 * 2);
    bf16_t* waw_bf  = (bf16_t*)alloc(128 * 256 * 2);
    bf16_t* wo_bf   = (bf16_t*)alloc(256 * 256 * 2);
    bf16_t* w1_bf   = (bf16_t*)alloc((size_t)DFF * 256 * 2);
    bf16_t* w2_bf   = (bf16_t*)alloc((size_t)256 * DFF * 2);
    // activations
    bf16_t* t2_bf   = (bf16_t*)alloc((size_t)Mrows * Cdim * 2);
    bf16_t* qk_bf   = (bf16_t*)alloc((size_t)Mrows * Cdim * 2);
    float*  qkv_f   = (float*)alloc((size_t)Mrows * 768 * 4);
    bf16_t* Qbf     = (bf16_t*)alloc((size_t)3 * BH * NQP * DH * 2); // Q,K,Vt contiguous
    bf16_t* Kbf     = Qbf + (size_t)BH * NQP * DH;
    bf16_t* Vtbf    = Kbf + (size_t)BH * NQP * DH;
    float*  scores  = (float*)alloc((size_t)BH * NQP * NQP * 4);
    bf16_t* attn_bf = (bf16_t*)alloc((size_t)BH * NQP * NQP * 2);
    float*  sa_f    = (float*)alloc((size_t)BH * NQP * DH * 4);
    bf16_t* sa_bf   = (bf16_t*)alloc((size_t)Mrows * Cdim * 2);
    float*  proj_f  = (float*)alloc((size_t)Mrows * Cdim * 4);
    float*  tgt1    = (float*)alloc((size_t)Mrows * Cdim * 4);
    bf16_t* t2b_bf  = (bf16_t*)alloc((size_t)Mrows * Cdim * 2);
    bf16_t* v_bf    = (bf16_t*)alloc((size_t)Bsz * LEN_IN * Cdim * 2);
    float*  off_f   = (float*)alloc((size_t)Mrows * 192 * 4);
    float*  aw_f    = (float*)alloc((size_t)Mrows * 96 * 4);
    float*  awsm_f  = (float*)alloc((size_t)Mrows * 96 * 4);
    bf16_t* ca_bf   = (bf16_t*)alloc((size_t)Mrows * Cdim * 2);
    float*  cap_f   = (float*)alloc((size_t)Mrows * Cdim * 4);
    float*  tgt2    = (float*)alloc((size_t)Mrows * Cdim * 4);
    bf16_t* t2c_bf  = (bf16_t*)alloc((size_t)Mrows * Cdim * 2);
    bf16_t* ffh_bf  = (bf16_t*)alloc((size_t)Mrows * DFF * 2);
    float*  ff2_f   = (float*)alloc((size_t)Mrows * Cdim * 4);

    auto cvt = [&](const float* s, bf16_t* dst, int n) {
        cvt_f32_bf16_kernel<<<(n + 255) / 256, 256, 0, stream>>>(s, dst, n);
    };
    // zero aw-weight pad rows, then fill valid rows
    zero_bf16_kernel<<<(128 * 256 + 255) / 256, 256, 0, stream>>>(waw_bf, 128 * 256);
    cvt(Wqkv, wqkv_bf, 768 * 256);
    cvt(Wout, wout_bf, 256 * 256);
    cvt(Wv,   wv_bf,   256 * 256);
    cvt(Woff, woff_bf, 192 * 256);
    cvt(Waw,  waw_bf,  96 * 256);
    cvt(Wo,   wo_bf,   256 * 256);
    cvt(W1,   w1_bf,   DFF * 256);
    cvt(W2,   w2_bf,   256 * DFF);

    const int NC = Mrows * Cdim;                 // 1228800

    // LN1: t2 = LN(tgt); qk = t2 + query_pos
    ln_kernel<<<Mrows, 256, 0, stream>>>(tgt, ln1g, ln1b, qpos, t2_bf, qk_bf);

    // zero padded Q/K/Vt
    {
        int n = 3 * BH * NQP * DH;
        zero_bf16_kernel<<<(n + 255) / 256, 256, 0, stream>>>(Qbf, n);
    }

    // fused QKV projection (Q,K from qk; V from t2) into qkv_f [4800,768]
    launch_gemm<4, GF_BIAS>(stream, qk_bf, 256, 0, wqkv_bf, 256, 0, bqkv,
                            qkv_f, 768, 0, Mrows, 512, 256, nullptr, 1);
    launch_gemm<4, GF_BIAS>(stream, t2_bf, 256, 0, wqkv_bf + (size_t)512 * 256, 256, 0,
                            bqkv + 512, qkv_f + 512, 768, 0, Mrows, 256, 256, nullptr, 1);

    split_qkv_kernel<<<(NC + 255) / 256, 256, 0, stream>>>(qkv_f, Qbf, Kbf, Vtbf);

    // scores = Q K^T (scale folded into Q), batched over B*H
    launch_gemm<4, 0>(stream, Qbf, DH, (long long)NQP * DH, Kbf, DH, (long long)NQP * DH,
                      nullptr, scores, NQP, (long long)NQP * NQP,
                      NQP, NQP, DH, nullptr, BH);

    attn_softmax_kernel<<<BH * NQP / 8, 256, 0, stream>>>(scores, attn_bf);

    // sa = attn @ V  (B operand = V^T stored [DH, NQP]); NT=2 exactly covers N=32
    launch_gemm<2, 0>(stream, attn_bf, NQP, (long long)NQP * NQP, Vtbf, NQP,
                      (long long)DH * NQP, nullptr, sa_f, DH, (long long)NQP * DH,
                      NQP, DH, NQP, nullptr, BH);

    merge_sa_kernel<<<(NC + 255) / 256, 256, 0, stream>>>(sa_f, sa_bf);

    // attention output projection + residual
    launch_gemm<4, GF_BIAS>(stream, sa_bf, 256, 0, wout_bf, 256, 0, bout,
                            proj_f, 256, 0, Mrows, 256, 256, nullptr, 1);
    add_f32_kernel<<<(NC + 255) / 256, 256, 0, stream>>>(tgt, proj_f, tgt1, NC);

    // LN2
    ln_kernel<<<Mrows, 256, 0, stream>>>(tgt1, ln2g, ln2b, nullptr, t2b_bf, nullptr);

    // value projection over memory (f32 A converted in-kernel), mask + bf16 out
    launch_gemm<4, GF_A_F32 | GF_BIAS | GF_OUT_BF16 | GF_MASK>(
        stream, mem, 256, 0, wv_bf, 256, 0, bv,
        v_bf, 256, 0, Bsz * LEN_IN, 256, 256, kmask, 1);

    // sampling offsets & weights
    launch_gemm<4, GF_BIAS>(stream, t2b_bf, 256, 0, woff_bf, 256, 0, boff,
                            off_f, 192, 0, Mrows, 192, 256, nullptr, 1);
    launch_gemm<4, GF_BIAS>(stream, t2b_bf, 256, 0, waw_bf, 256, 0, baw,
                            aw_f, 96, 0, Mrows, 96, 256, nullptr, 1);
    aw_softmax_kernel<<<(Mrows * Hh + 255) / 256, 256, 0, stream>>>(aw_f, awsm_f);

    deform_sample_kernel<<<Mrows, 256, 0, stream>>>(off_f, awsm_f, qrp, v_bf, ca_bf);

    // output projection + residual
    launch_gemm<4, GF_BIAS>(stream, ca_bf, 256, 0, wo_bf, 256, 0, bo,
                            cap_f, 256, 0, Mrows, 256, 256, nullptr, 1);
    add_f32_kernel<<<(NC + 255) / 256, 256, 0, stream>>>(tgt1, cap_f, tgt2, NC);

    // LN3 + FFN
    ln_kernel<<<Mrows, 256, 0, stream>>>(tgt2, ln3g, ln3b, nullptr, t2c_bf, nullptr);
    launch_gemm<4, GF_BIAS | GF_RELU | GF_OUT_BF16>(
        stream, t2c_bf, 256, 0, w1_bf, 256, 0, b1,
        ffh_bf, DFF, 0, Mrows, DFF, 256, nullptr, 1);
    launch_gemm<4, GF_BIAS>(stream, ffh_bf, DFF, 0, w2_bf, DFF, 0, b2,
                            ff2_f, 256, 0, Mrows, 256, DFF, nullptr, 1);

    add_f32_kernel<<<(NC + 255) / 256, 256, 0, stream>>>(tgt2, ff2_f, (float*)d_out, NC);
}
